// BasicBlock_62972810494326
// MI455X (gfx1250) — matrix operand order repaired and checked by reading, not compiled, not run on hardware
//
#include <hip/hip_runtime.h>
#include <hip/hip_bf16.h>

// ---------------------------------------------------------------------------
// MI455X (gfx1250, wave32) implementation.
// Heavy 32->32 3x3 dilated convs run as implicit GEMM on v_wmma_f32_16x16x32_bf16:
//   M = 16 pixels (one wave row), N = 16 output channels (x2 tiles), K = 32 in-ch per tap.
// Per-channel dilation selection / argsort permutation / mul scaling from the
// reference's adconv are folded into the B fragments (zeroed columns per dilation).
// Activations between adconvs live in bf16 NHWC [b][y][x][32]; boundary math fp32.
// Interior tiles are staged LDS<-global by the Tensor Data Mover (TENSORcnt).
// ---------------------------------------------------------------------------

typedef __attribute__((ext_vector_type(16))) __bf16 v16bf;
typedef __attribute__((ext_vector_type(8)))  float  v8f;
typedef __attribute__((ext_vector_type(4)))  unsigned int v4u;
typedef __attribute__((ext_vector_type(8)))  int  v8i;
typedef __attribute__((ext_vector_type(4)))  int  v4i;

union AFrag { v16bf v; uint4 q[2]; };
union CAcc  { v8f v; float f[8]; };

__device__ __forceinline__ unsigned short f2bf(float f) {
  union { float f; unsigned int u; } x; x.f = f;
  unsigned int u = x.u;
  unsigned int r = (u + 0x7FFFu + ((u >> 16) & 1u)) >> 16;  // RNE
  return (unsigned short)r;
}
__device__ __forceinline__ float bf2f(unsigned short h) {
  union { unsigned int u; float f; } x; x.u = ((unsigned int)h) << 16;
  return x.f;
}

#define IMGH 320
#define IMGW 320
#define NB   4

// ---------------------------------------------------------------------------
// Per-adconv-layer metadata: src[j] (permuted source channel), dil[j] (1..3),
// scale[j] = mul[idx[src[j]]]. Matches argmax -> stable argsort -> mul of ref.
// Layer layout in ws: 32 int src | 32 int dil | 32 float scale  (96 words)
// ---------------------------------------------------------------------------
__global__ void meta_kernel(const float* c1f, const float* c2f, const float* c1b,
                            const float* c2b, const float* c1g, const float* c2g,
                            int* meta) {
  if (threadIdx.x != 0) return;
  int L = blockIdx.x;
  const float* p; int T;
  switch (L) {
    case 0: p = c1f; T = 1; break;
    case 1: p = c2f; T = 3; break;
    case 2: p = c1b; T = 1; break;
    case 3: p = c2b; T = 3; break;
    case 4: p = c1g; T = 1; break;
    default: p = c2g; T = 3; break;
  }
  int* src = meta + L * 96;
  int* dil = src + 32;
  float* scl = (float*)(dil + 32);
  int idx[32]; float sum[3]; int cnt[3];
  for (int t = 0; t < 3; t++) { sum[t] = 0.0f; cnt[t] = 0; }
  for (int c = 0; c < 32; c++) {
    int best = 0; float bv = p[c * T];
    for (int t = 1; t < T; t++) { float v = p[c * T + t]; if (v > bv) { bv = v; best = t; } }
    idx[c] = best; sum[best] += p[c * T + best]; cnt[best]++;
  }
  float mul[3];
  for (int t = 0; t < T; t++) mul[t] = sum[t] / fmaxf((float)cnt[t], 1.0f) + 1.0f;
  int j = 0;
  for (int t = 0; t < T; t++)             // stable argsort by dilation index
    for (int c = 0; c < 32; c++)
      if (idx[c] == t) { src[j] = c; dil[j] = t + 1; scl[j] = mul[t]; j++; }
}

// ---------------------------------------------------------------------------
// phi = stride-32 conv of x (and deltax = x - xprev) with PhiW[:, c].
// One block per (patch, channel, batch); patch pixels cached in LDS.
// phi layout: [b][c][patch][m] (m contiguous, 256).
// ---------------------------------------------------------------------------
__global__ __launch_bounds__(256) void phi_kernel(
    const float* __restrict__ x, const float* __restrict__ xprev,
    const float* __restrict__ PhiW,
    float* __restrict__ phix, float* __restrict__ phidx) {
  __shared__ float sx[1024], sdx[1024];
  int patch = blockIdx.x, c = blockIdx.y, b = blockIdx.z;
  int py = patch / 10, px = patch % 10;
  size_t base = (((size_t)b * 3 + c) * IMGH + py * 32) * IMGW + px * 32;
  for (int i = threadIdx.x; i < 1024; i += 256) {
    int iy = i >> 5, ix = i & 31;
    size_t g = base + (size_t)iy * IMGW + ix;
    float xv = x[g];
    sx[i] = xv; sdx[i] = xv - xprev[g];
  }
  __syncthreads();
  int m = threadIdx.x;
  const float* wr = PhiW + ((size_t)m * 3 + c) * 1024;
  float a = 0.0f, d = 0.0f;
  for (int i = 0; i < 1024; i++) { float w = wr[i]; a += w * sx[i]; d += w * sdx[i]; }
  size_t o = (((size_t)b * 3 + c) * 100 + patch) * 256 + m;
  phix[o] = a; phidx[o] = d;
}

// ---------------------------------------------------------------------------
// zeta = x - lam*PhiTPhi(x) - lam*xi*PhiTPhi(dx) + xi*dx + lam*PhiTb  (NCHW f32)
// PhiT (1x1) + pixel_shuffle fused: out pixel (py*32+iy, px*32+ix) uses
// PhiTW row (c*1024 + iy*32+ix) dotted with phi[b][c][patch][:256].
// ---------------------------------------------------------------------------
__global__ __launch_bounds__(256) void zeta_kernel(
    const float* __restrict__ x, const float* __restrict__ xprev,
    const float* __restrict__ PhiTW, const float* __restrict__ PhiTb,
    const float* __restrict__ phix, const float* __restrict__ phidx,
    const float* lam_p, const float* t_p, float* __restrict__ zeta) {
  __shared__ float sxm[256], sdm[256];
  int patch = blockIdx.x, c = blockIdx.y, b = blockIdx.z;
  size_t po = (((size_t)b * 3 + c) * 100 + patch) * 256;
  sxm[threadIdx.x] = phix[po + threadIdx.x];
  sdm[threadIdx.x] = phidx[po + threadIdx.x];
  __syncthreads();
  float lam = *lam_p, tt = *t_p;
  float tplus = (1.0f + sqrtf(1.0f + 4.0f * tt * tt)) * 0.5f;
  float xi = (tt - 1.0f) / tplus;
  int py = patch / 10, px = patch % 10;
  for (int n = threadIdx.x; n < 1024; n += 256) {
    int iy = n >> 5, ix = n & 31;
    const float* wr = PhiTW + (size_t)(c * 1024 + n) * 256;
    float a = 0.0f, d = 0.0f;
    for (int m2 = 0; m2 < 256; m2++) { float w = wr[m2]; a += w * sxm[m2]; d += w * sdm[m2]; }
    size_t g = (((size_t)b * 3 + c) * IMGH + py * 32 + iy) * IMGW + px * 32 + ix;
    float xv = x[g], dv = xv - xprev[g];
    zeta[g] = xv - lam * a - lam * xi * d + xi * dv + lam * PhiTb[g];
  }
}

// ---------------------------------------------------------------------------
// conv_D: 3->32, 3x3, pad 1.  Writes act (bf16 NHWC) and x_D (f32 NCHW, parked
// in the symloss region of d_out so the final adconv can fuse x_D_est - x_D).
// ---------------------------------------------------------------------------
__global__ __launch_bounds__(256) void convD_kernel(
    const float* __restrict__ zeta, const float* __restrict__ wD,
    unsigned short* __restrict__ act, float* __restrict__ xD) {
  int pid = blockIdx.x * 256 + threadIdx.x;
  int xc = pid % IMGW; int y = (pid / IMGW) % IMGH; int b = pid / (IMGW * IMGH);
  if (b >= NB) return;
  float inv[3][3][3];
  for (int ci = 0; ci < 3; ci++)
    for (int dy = 0; dy < 3; dy++)
      for (int dx = 0; dx < 3; dx++) {
        int yy = y + dy - 1, xx = xc + dx - 1;
        inv[ci][dy][dx] = (yy >= 0 && yy < IMGH && xx >= 0 && xx < IMGW)
            ? zeta[(((size_t)b * 3 + ci) * IMGH + yy) * IMGW + xx] : 0.0f;
      }
  size_t abase = (((size_t)b * IMGH + y) * IMGW + xc) * 32;
  for (int o = 0; o < 32; o++) {
    float acc = 0.0f;
    const float* wr = wD + o * 27;
    for (int ci = 0; ci < 3; ci++)
      for (int dy = 0; dy < 3; dy++)
        for (int dx = 0; dx < 3; dx++)
          acc += wr[(ci * 3 + dy) * 3 + dx] * inv[ci][dy][dx];
    act[abase + o] = f2bf(acc);
    xD[(((size_t)b * 32 + o) * IMGH + y) * IMGW + xc] = acc;
  }
}

// ---------------------------------------------------------------------------
// adconv via WMMA bf16.  Block = 256 threads = 8 waves; wave w computes output
// row y0+w, 16 columns, all 32 channels.  Input tile (14x22x32 bf16, halo 3 for
// dilation<=3) staged in LDS -- by the Tensor Data Mover (single 2D descriptor,
// rows contiguous in NHWC) for interior tiles with no pre-op, else by a manual
// zero-padded/pre-op'd cooperative copy.  B fragments per tap (all dilations,
// both N-halves) are built in LDS with dilation-select/permute/scale folded in.
// preop/postop: 0=none 1=relu 2=soft-threshold.  outmode: 0=bf16 NHWC,
// 1 = f32 NCHW with out = acc - out (symloss fusion).
// ---------------------------------------------------------------------------
__global__ __launch_bounds__(256) void adconv_wmma_kernel(
    const unsigned short* __restrict__ in, const float* __restrict__ w,
    const int* __restrict__ meta, int T, int preop, int postop, int outmode,
    unsigned short* __restrict__ outb, float* __restrict__ outf,
    const float* __restrict__ thr_p) {
  __shared__ __align__(16) unsigned short tileS[14 * 22 * 32];
  __shared__ __align__(16) unsigned short Bsh[3 * 2 * 32 * 16];
  const int* src = meta;
  const int* dil = meta + 32;
  const float* scl = (const float*)(meta + 64);
  int tid = threadIdx.x;
  int lane = tid & 31, wv = tid >> 5;
  int x0 = blockIdx.x * 16, y0 = blockIdx.y * 8, b = blockIdx.z;
  float thr = *thr_p;

  __builtin_prefetch(w, 0, 3);  // warm weights toward L2 (global_prefetch_b8)

  // ---- Stage input tile (halo 3) into LDS ------------------------------
  bool staged = false;
#if __has_builtin(__builtin_amdgcn_tensor_load_to_lds) && __has_builtin(__builtin_amdgcn_s_wait_tensorcnt)
  // Interior tile, no pre-op: one TDM 2D-descriptor DMA moves the whole
  // 14 x (22*32) bf16 tile (rows contiguous in NHWC) into LDS.
  bool tdm_ok = (preop == 0) &&
                (x0 - 3 >= 0) && (x0 + 18 <= IMGW - 1) &&
                (y0 - 3 >= 0) && (y0 + 10 <= IMGH - 1);
  if (tdm_ok) {
    if (wv == 0) {
      unsigned lds_base = (unsigned)(uintptr_t)(&tileS[0]);  // flat[31:0] == LDS offset
      unsigned long long ga = (unsigned long long)(uintptr_t)
          (in + (((size_t)b * IMGH + (y0 - 3)) * IMGW + (x0 - 3)) * 32);
      // D# group 0: count=1 | lds_addr | global_addr | type=2
      v4u g0;
      g0.x = 1u;
      g0.y = lds_base;
      g0.z = (unsigned)(ga & 0xFFFFFFFFu);
      g0.w = (unsigned)((ga >> 32) & 0x01FFFFFFu) | 0x80000000u;
      // D# group 1: data_size=1 (2B) | tensor_dim0/1 huge (no OOB clip) |
      // tile_dim0=704 elts, tile_dim1=14 rows | dim0 stride=320*32 elts
      unsigned td0 = 0x40000000u;      // tensor_dim0 (elements)
      unsigned td1 = 0x00004000u;      // tensor_dim1 (rows)
      unsigned tile0 = 22u * 32u;      // 704 elements per tile row
      unsigned tile1 = 14u;            // rows
      unsigned stride0 = (unsigned)IMGW * 32u;  // 10240 elements
      v8i g1;
      g1[0] = 0x00010000;                                   // data_size=1 (2B)
      g1[1] = (int)((td0 & 0xFFFFu) << 16);                 // [63:48] td0 lo16
      g1[2] = (int)((td0 >> 16) | ((td1 & 0xFFFFu) << 16)); // td0 hi16 | td1 lo16
      g1[3] = (int)((td1 >> 16) | (tile0 << 16));           // td1 hi16 | tile_dim0
      g1[4] = (int)(tile1);                                 // tile_dim1 | tile_dim2=0
      g1[5] = (int)(stride0);                               // dim0 stride lo32
      g1[6] = 0;                                            // stride hi | d1stride lo
      g1[7] = 0;
      v4i g2 = {0, 0, 0, 0}, g3 = {0, 0, 0, 0};             // 2D: groups 2/3 unused
      v8i g4 = {0, 0, 0, 0, 0, 0, 0, 0};                    // trailing group (6-arg form)
      __builtin_amdgcn_tensor_load_to_lds(g0, g1, g2, g3, g4, 0);
      __builtin_amdgcn_s_wait_tensorcnt(0);                 // TENSORcnt == 0
    }
    staged = true;
  }
#endif
  if (!staged) {
    for (int p = tid; p < 14 * 22; p += 256) {
      int py = p / 22, px = p % 22;
      int gy = y0 - 3 + py, gx = x0 - 3 + px;
      unsigned short* d = &tileS[p * 32];
      if (gy >= 0 && gy < IMGH && gx >= 0 && gx < IMGW) {
        const unsigned short* s = in + (((size_t)b * IMGH + gy) * IMGW + gx) * 32;
        if (preop == 0) {
          const uint4* s4 = (const uint4*)s; uint4* d4 = (uint4*)d;
          for (int q = 0; q < 4; q++) d4[q] = s4[q];
        } else {
          for (int c = 0; c < 32; c++) {
            float v = bf2f(s[c]);
            v = (preop == 1) ? fmaxf(v, 0.0f)
                             : copysignf(fmaxf(fabsf(v) - thr, 0.0f), v);
            d[c] = f2bf(v);
          }
        }
      } else {
        uint4 z = {0, 0, 0, 0}; uint4* d4 = (uint4*)d;
        for (int q = 0; q < 4; q++) d4[q] = z;
      }
    }
  }

  CAcc acc0, acc1;
  for (int i = 0; i < 8; i++) { acc0.f[i] = 0.0f; acc1.f[i] = 0.0f; }
  __syncthreads();

  int m = lane & 15, kh = lane >> 4;
  for (int tap = 0; tap < 9; tap++) {
    int ty = tap / 3, tx = tap % 3;
    // Build B fragments for this tap: per dilation t, column j holds
    // scale[j]*W[src[j], K, tap] iff dil[j]==t+1, else 0.
    // B(32x16) layout: lane holds col n=lane&15, K = (lane>>4)*16 + i.
    for (int e = tid; e < T * 1024; e += 256) {
      int i = e & 15; int bl = (e >> 4) & 31; int nh = (e >> 9) & 1; int t = e >> 10;
      int K = (bl >> 4) * 16 + i;
      int j = (bl & 15) + nh * 16;
      float val = 0.0f;
      if (dil[j] - 1 == t)
        val = w[((size_t)(src[j] * 32 + K)) * 9 + tap] * scl[j];
      Bsh[((t * 2 + nh) * 32 + bl) * 16 + i] = f2bf(val);
    }
    __syncthreads();
    for (int t = 0; t < T; t++) {
      int d = t + 1;
      int py = 3 + wv + (ty - 1) * d;
      int px = 3 + m + (tx - 1) * d;
      // A(16x32) layout: lane holds row M=lane&15; kh=0 -> K 0..15,
      // kh=1 -> K 16..31  (K = input channel).
      const unsigned short* pix = &tileS[(py * 22 + px) * 32];
      AFrag a;
      a.q[0] = *(const uint4*)(pix + kh * 16);
      a.q[1] = *(const uint4*)(pix + kh * 16 + 8);
      AFrag b0, b1;
      const unsigned short* bp0 = &Bsh[((t * 2 + 0) * 32 + lane) * 16];
      b0.q[0] = *(const uint4*)(bp0);
      b0.q[1] = *(const uint4*)(bp0 + 8);
      const unsigned short* bp1 = &Bsh[((t * 2 + 1) * 32 + lane) * 16];
      b1.q[0] = *(const uint4*)(bp1);
      b1.q[1] = *(const uint4*)(bp1 + 8);
      acc0.v = __builtin_amdgcn_wmma_f32_16x16x32_bf16(
          false, a.v, false, b0.v, (short)0, acc0.v, false, false);
      acc1.v = __builtin_amdgcn_wmma_f32_16x16x32_bf16(
          false, a.v, false, b1.v, (short)0, acc1.v, false, false);
    }
    __syncthreads();  // Bsh rewritten next tap
  }

  // C/D layout: VGPR r -> row M = r + 8*(lane>=16); col N = lane&15.
  int y = y0 + wv;
  for (int nh = 0; nh < 2; nh++) {
    CAcc* A = nh ? &acc1 : &acc0;
    int ch = (lane & 15) + nh * 16;
    for (int r = 0; r < 8; r++) {
      int mm = r + kh * 8;
      int xx = x0 + mm;
      float v = A->f[r];
      if (postop == 1) v = fmaxf(v, 0.0f);
      else if (postop == 2) v = copysignf(fmaxf(fabsf(v) - thr, 0.0f), v);
      if (outmode == 0) {
        outb[(((size_t)b * IMGH + y) * IMGW + xx) * 32 + ch] = f2bf(v);
      } else {
        size_t g = (((size_t)b * 32 + ch) * IMGH + y) * IMGW + xx;
        outf[g] = v - outf[g];  // x_D_est - x_D (x_D parked here by convD)
      }
    }
  }
}

// ---------------------------------------------------------------------------
// conv3_G (32->3, 3x3, pad 1) + x_pred = zeta + mergeScale * x_G  (f32 NCHW)
// ---------------------------------------------------------------------------
__global__ __launch_bounds__(256) void convG_kernel(
    const unsigned short* __restrict__ h, const float* __restrict__ wG,
    const float* __restrict__ zeta, const float* ms_p, float* __restrict__ pred) {
  int pid = blockIdx.x * 256 + threadIdx.x;
  int xc = pid % IMGW; int y = (pid / IMGW) % IMGH; int b = pid / (IMGW * IMGH);
  if (b >= NB) return;
  float acc[3] = {0.0f, 0.0f, 0.0f};
  for (int dy = 0; dy < 3; dy++) {
    int yy = y + dy - 1; if (yy < 0 || yy >= IMGH) continue;
    for (int dx = 0; dx < 3; dx++) {
      int xx = xc + dx - 1; if (xx < 0 || xx >= IMGW) continue;
      const unsigned short* hp = h + (((size_t)b * IMGH + yy) * IMGW + xx) * 32;
      for (int ci = 0; ci < 32; ci++) {
        float v = bf2f(hp[ci]);
        for (int o = 0; o < 3; o++)
          acc[o] += wG[((o * 32 + ci) * 3 + dy) * 3 + dx] * v;
      }
    }
  }
  float ms = *ms_p;
  for (int o = 0; o < 3; o++) {
    size_t g = (((size_t)b * 3 + o) * IMGH + y) * IMGW + xc;
    pred[g] = zeta[g] + acc[o] * ms;
  }
}

// ---------------------------------------------------------------------------
extern "C" void kernel_launch(void* const* d_in, const int* in_sizes, int n_in,
                              void* d_out, int out_size, void* d_ws, size_t ws_size,
                              hipStream_t stream) {
  const float* xprev = (const float*)d_in[0];
  const float* x     = (const float*)d_in[1];
  const float* PhiW  = (const float*)d_in[2];
  const float* PhiTW = (const float*)d_in[3];
  const float* PhiTb = (const float*)d_in[4];
  const float* lam   = (const float*)d_in[5];
  const float* sthr  = (const float*)d_in[6];
  const float* tpar  = (const float*)d_in[7];
  const float* msc   = (const float*)d_in[8];
  const float* wD    = (const float*)d_in[9];
  const float* wG    = (const float*)d_in[10];
  const float* c1f_w = (const float*)d_in[11];
  const float* c1f_p = (const float*)d_in[12];
  const float* c2f_w = (const float*)d_in[13];
  const float* c2f_p = (const float*)d_in[14];
  const float* c1b_w = (const float*)d_in[15];
  const float* c1b_p = (const float*)d_in[16];
  const float* c2b_w = (const float*)d_in[17];
  const float* c2b_p = (const float*)d_in[18];
  const float* c1g_w = (const float*)d_in[19];
  const float* c1g_p = (const float*)d_in[20];
  const float* c2g_w = (const float*)d_in[21];
  const float* c2g_p = (const float*)d_in[22];
  (void)in_sizes; (void)n_in; (void)out_size; (void)ws_size;

  float* out  = (float*)d_out;
  float* pred = out;                                    // 4*3*320*320
  float* sym  = out + (size_t)NB * 3 * IMGH * IMGW;     // 4*32*320*320

  char* ws = (char*)d_ws;
  size_t phiSz  = (size_t)NB * 3 * 100 * 256 * 4;
  size_t zetaSz = (size_t)NB * 3 * IMGH * IMGW * 4;
  size_t actSz  = (size_t)NB * IMGH * IMGW * 32 * 2;    // bf16 NHWC
  float* phix  = (float*)ws;
  float* phidx = (float*)(ws + phiSz);
  float* zeta  = (float*)(ws + 2 * phiSz);
  int*   meta  = (int*)(ws + 2 * phiSz + zetaSz);
  char*  acts  = ws + 2 * phiSz + zetaSz + 4096;
  unsigned short* actA = (unsigned short*)acts;
  unsigned short* actB = (unsigned short*)(acts + actSz);
  unsigned short* actC = (unsigned short*)(acts + 2 * actSz);

  meta_kernel<<<6, 32, 0, stream>>>(c1f_p, c2f_p, c1b_p, c2b_p, c1g_p, c2g_p, meta);

  dim3 gphi(100, 3, NB);
  phi_kernel<<<gphi, 256, 0, stream>>>(x, xprev, PhiW, phix, phidx);
  zeta_kernel<<<gphi, 256, 0, stream>>>(x, xprev, PhiTW, PhiTb, phix, phidx, lam, tpar, zeta);

  int npix = NB * IMGH * IMGW;
  convD_kernel<<<(npix + 255) / 256, 256, 0, stream>>>(zeta, wD, actA, sym);

  dim3 gad(IMGW / 16, IMGH / 8, NB);  // (20, 40, 4)
  // h1 = relu(adconv(x_D, c1f))
  adconv_wmma_kernel<<<gad, 256, 0, stream>>>(actA, c1f_w, meta + 0 * 96, 1, 0, 1, 0, actB, nullptr, sthr);
  // x_forward = adconv(h1, c2f)   (kept in actC)
  adconv_wmma_kernel<<<gad, 256, 0, stream>>>(actB, c2f_w, meta + 1 * 96, 3, 0, 0, 0, actC, nullptr, sthr);
  // h2 = relu(adconv(softthr(x_forward), c1f))
  adconv_wmma_kernel<<<gad, 256, 0, stream>>>(actC, c1f_w, meta + 0 * 96, 1, 2, 1, 0, actB, nullptr, sthr);
  // x_backward = adconv(h2, c2b)
  adconv_wmma_kernel<<<gad, 256, 0, stream>>>(actB, c2b_w, meta + 3 * 96, 3, 0, 0, 0, actA, nullptr, sthr);
  // h3 = adconv(relu(x_backward), c1g)
  adconv_wmma_kernel<<<gad, 256, 0, stream>>>(actA, c1g_w, meta + 4 * 96, 1, 1, 0, 0, actB, nullptr, sthr);
  // h4 = adconv(relu(h3), c2g)
  adconv_wmma_kernel<<<gad, 256, 0, stream>>>(actB, c2g_w, meta + 5 * 96, 3, 1, 0, 0, actA, nullptr, sthr);
  // x_pred = zeta + mergeScale * conv3_G(h4)
  convG_kernel<<<(npix + 255) / 256, 256, 0, stream>>>(actA, wG, zeta, msc, pred);
  // h5 = relu(adconv(x_forward, c1b))
  adconv_wmma_kernel<<<gad, 256, 0, stream>>>(actC, c1b_w, meta + 2 * 96, 1, 0, 1, 0, actB, nullptr, sthr);
  // symloss = adconv(h5, c2b) - x_D   (x_D parked in sym by convD)
  adconv_wmma_kernel<<<gad, 256, 0, stream>>>(actB, c2b_w, meta + 3 * 96, 3, 0, 0, 1, nullptr, sym, sthr);
}
